// CrossNegativeAttention_34273839022121
// MI455X (gfx1250) — compile-verified, matrix operand order checked by reading, not run on hardware
//
#include <hip/hip_runtime.h>
#include <hip/hip_bf16.h>

// ---------------------------------------------------------------------------
// CDNA5 / gfx1250 implementation of CrossNegativeAttention.
// All matmuls lowered to v_wmma_f32_16x16x32_f16 (wave32).
// Fragment-layout assumptions per CDNA5 ISA §7.12.2:
//   A (16x32 f16): lane -> M (lane&15), elems j=0..7 -> K = (lane>>4)*8 + j,
//                  j=8..15 -> K = 16 + (lane>>4)*8 + (j-8)
//   B loaded from row-major [N,K] with the same indexing (lane -> N).
//   C/D (16x16 f32): lane -> N (lane&15), VGPR i -> M = i + (lane>>4)*8
// ---------------------------------------------------------------------------

typedef __attribute__((ext_vector_type(16))) _Float16 v16h;
typedef __attribute__((ext_vector_type(8)))  _Float16 v8h;
typedef __attribute__((ext_vector_type(8)))  float    v8f;

constexpr int Bc = 2;
constexpr int Cc = 1024;
constexpr int Tc = 2048;
constexpr int Hh = 16;
constexpr int Dd = 64;
constexpr float SCALE = 0.125f;   // 1/sqrt(64)
#define NEG_INF (-__builtin_inff())

__device__ inline v8f wmma16(v16h a, v16h b, v8f c) {
  // D = A(16x32) * B(32x16) + C ; emits v_wmma_f32_16x16x32_f16
  return __builtin_amdgcn_wmma_f32_16x16x32_f16(
      /*neg_a=*/false, a, /*neg_b=*/false, b,
      /*c_mod=*/(short)0, c, /*reuse_a=*/false, /*reuse_b=*/false);
}

// Load a 16x32 f16 fragment (A-operand layout) from a row-major [rows, ld]
// matrix at (row0, k0). Also used for B-operands stored as row-major [N, K].
__device__ inline v16h load_frag(const _Float16* __restrict__ M,
                                 int row0, int k0, int ld) {
  int lane = threadIdx.x & 31;
  const _Float16* p = M + (size_t)(row0 + (lane & 15)) * ld + k0 + ((lane >> 4) << 3);
  v8h lo = *(const v8h*)(p);        // K offsets half*8 + 0..7
  v8h hi = *(const v8h*)(p + 16);   // K offsets 16 + half*8 + 0..7
  v16h f;
#pragma unroll
  for (int j = 0; j < 8; ++j) { f[j] = lo[j]; f[j + 8] = hi[j]; }
  return f;
}

// ---------------------------------------------------------------------------
// Stage 0: f32 -> f16 weight conversion
// ---------------------------------------------------------------------------
__global__ void f32_to_f16_kernel(const float* __restrict__ in,
                                  _Float16* __restrict__ out, int n) {
  int i = blockIdx.x * blockDim.x + threadIdx.x;
  if (i < n) out[i] = (_Float16)in[i];
}

// ---------------------------------------------------------------------------
// Stage 1: depthwise conv3 (pad 1) + timestep mask + LayerNorm over channels.
// One block per (b,t) column, blockIdx.y selects {q,k,v} branch.
// Output: f16 row-major [B*T, C]  (ready as WMMA A-operand, K = C).
// ---------------------------------------------------------------------------
__global__ __launch_bounds__(256)
void conv_ln_kernel(const float* __restrict__ xq, const float* __restrict__ xkv,
                    const float* __restrict__ wq, const float* __restrict__ wk,
                    const float* __restrict__ wv,
                    const float* __restrict__ gq, const float* __restrict__ beq,
                    const float* __restrict__ gk, const float* __restrict__ bek,
                    const float* __restrict__ gv, const float* __restrict__ bev,
                    const unsigned char* __restrict__ mask,
                    _Float16* __restrict__ oq, _Float16* __restrict__ ok,
                    _Float16* __restrict__ ov) {
  int bt = blockIdx.x;
  int z  = blockIdx.y;
  int b  = bt / Tc, t = bt % Tc;
  const float* x  = (z == 0) ? xq : xkv;
  const float* w  = (z == 0) ? wq : (z == 1) ? wk : wv;
  const float* g  = (z == 0) ? gq : (z == 1) ? gk : gv;
  const float* be = (z == 0) ? beq : (z == 1) ? bek : bev;
  _Float16* o     = (z == 0) ? oq : (z == 1) ? ok : ov;
  float mf = mask[bt] ? 1.f : 0.f;
  int tid = threadIdx.x;

  float y[4];
  float s = 0.f, s2 = 0.f;
#pragma unroll
  for (int j = 0; j < 4; ++j) {
    int c = tid + j * 256;
    const float* xp = x + ((size_t)b * Cc + c) * Tc;
    float xm = (t > 0)      ? xp[t - 1] : 0.f;
    float x0 = xp[t];
    float xp1 = (t < Tc - 1) ? xp[t + 1] : 0.f;
    float v = (w[c * 3 + 0] * xm + w[c * 3 + 1] * x0 + w[c * 3 + 2] * xp1) * mf;
    y[j] = v; s += v; s2 += v * v;
  }
  __shared__ float red0[256];
  __shared__ float red1[256];
  red0[tid] = s; red1[tid] = s2;
  __syncthreads();
  for (int st = 128; st > 0; st >>= 1) {
    if (tid < st) { red0[tid] += red0[tid + st]; red1[tid] += red1[tid + st]; }
    __syncthreads();
  }
  float mean = red0[0] * (1.f / Cc);
  float var  = red1[0] * (1.f / Cc) - mean * mean;
  float rstd = rsqrtf(var + 1e-5f);
#pragma unroll
  for (int j = 0; j < 4; ++j) {
    int c = tid + j * 256;
    float val = (y[j] - mean) * rstd * g[c] + be[c];
    o[(size_t)bt * Cc + c] = (_Float16)val;
  }
}

// ---------------------------------------------------------------------------
// Stage 2/4: GEMM  out[m,n] = sum_k A[m,k] * W[n,k] + bias[n]
// One wave computes a 32x32 tile (4 WMMAs per 32-wide K step).
// mode 0: f16 out row-major [M,N]                       (q, k projections)
// mode 1: f16 out transposed [b, n, t], * mask[t]       (v projection)
// mode 2: f32 out [b, n, t] = resid + mask*(acc+bias)   (final projection)
// ---------------------------------------------------------------------------
__global__ __launch_bounds__(128)
void gemm_wmma_kernel(const _Float16* __restrict__ A,
                      const _Float16* __restrict__ W,
                      const float* __restrict__ bias,
                      int M, int N, int K, int mode,
                      const unsigned char* __restrict__ mask,
                      const float* __restrict__ resid,
                      _Float16* __restrict__ out16,
                      float* __restrict__ out32) {
  int wave = threadIdx.x >> 5;
  int lane = threadIdx.x & 31;
  int half = lane >> 4;
  int n0 = blockIdx.x * 32;
  int m0 = (blockIdx.y * 4 + wave) * 32;
  if (m0 >= M || n0 >= N) return;

  v8f a00 = {}, a01 = {}, a10 = {}, a11 = {};
  for (int k0 = 0; k0 < K; k0 += 32) {
    v16h fa0 = load_frag(A, m0,      k0, K);
    v16h fa1 = load_frag(A, m0 + 16, k0, K);
    v16h fb0 = load_frag(W, n0,      k0, K);
    v16h fb1 = load_frag(W, n0 + 16, k0, K);
    a00 = wmma16(fa0, fb0, a00);
    a01 = wmma16(fa0, fb1, a01);
    a10 = wmma16(fa1, fb0, a10);
    a11 = wmma16(fa1, fb1, a11);
  }

#pragma unroll
  for (int mt = 0; mt < 2; ++mt) {
#pragma unroll
    for (int nt = 0; nt < 2; ++nt) {
      v8f acc = (mt == 0) ? ((nt == 0) ? a00 : a01) : ((nt == 0) ? a10 : a11);
      int n = n0 + nt * 16 + (lane & 15);
      float bv = bias[n];
#pragma unroll
      for (int i = 0; i < 8; ++i) {
        int m = m0 + mt * 16 + half * 8 + i;
        float val = acc[i] + bv;
        if (mode == 0) {
          out16[(size_t)m * N + n] = (_Float16)val;
        } else if (mode == 1) {
          int b = m / Tc, t = m % Tc;
          float mf = mask[b * Tc + t] ? 1.f : 0.f;
          out16[(size_t)b * N * Tc + (size_t)n * Tc + t] = (_Float16)(val * mf);
        } else {
          int b = m / Tc, t = m % Tc;
          float mf = mask[b * Tc + t] ? 1.f : 0.f;
          size_t idx = (size_t)b * N * Tc + (size_t)n * Tc + t;
          out32[idx] = resid[idx] + mf * val;
        }
      }
    }
  }
}

// ---------------------------------------------------------------------------
// Stage 3: fused negated-softmax attention (flash style, online softmax).
// One wave per 16-row q tile per (b,h). Computes S^T = K*Q^T so the f32 D
// fragment maps directly (per lane) onto the f16 A-fragment of P for P*V.
// Row statistics (lane = t) broadcast to O rows via ds_bpermute (__shfl).
// ---------------------------------------------------------------------------
__global__ __launch_bounds__(32)
void attn_wmma_kernel(const _Float16* __restrict__ Q,   // [B*T, C]
                      const _Float16* __restrict__ Kp,  // [B*T, C]
                      const _Float16* __restrict__ Vt,  // [B, C, T]
                      const unsigned char* __restrict__ mask,
                      _Float16* __restrict__ Out) {     // [B*T, C]
  int lane = threadIdx.x & 31;
  int half = lane >> 4;
  int t0 = blockIdx.x * 16;
  int bh = blockIdx.y;
  int b = bh / Hh, h = bh % Hh;
  const _Float16* qb = Q  + (size_t)b * Tc * Cc + h * Dd;       // ld = C
  const _Float16* kb = Kp + (size_t)b * Tc * Cc + h * Dd;       // ld = C
  const _Float16* vb = Vt + (size_t)b * Cc * Tc + (size_t)h * Dd * Tc; // ld = T
  const unsigned char* mb = mask + b * Tc;

  // Q^T as B-operand (K = d), loaded once per q tile.
  v16h bq0 = load_frag(qb, t0, 0,  Cc);
  v16h bq1 = load_frag(qb, t0, 32, Cc);

  v8f o0 = {}, o1 = {}, o2 = {}, o3 = {};
  float m_run = NEG_INF;
  float ell = 0.f;

  for (int s0 = 0; s0 < Tc; s0 += 32) {
    // S^T tiles: rows s (M), cols t (N); K = d = 64 -> 2 WMMAs each.
    v8f st0 = {}, st1 = {};
    st0 = wmma16(load_frag(kb, s0,      0,  Cc), bq0, st0);
    st0 = wmma16(load_frag(kb, s0,      32, Cc), bq1, st0);
    st1 = wmma16(load_frag(kb, s0 + 16, 0,  Cc), bq0, st1);
    st1 = wmma16(load_frag(kb, s0 + 16, 32, Cc), bq1, st1);

    // Masked, scaled logits. Lane holds column t = lane&15;
    // st0 rows: s = s0 + half*8 + i ; st1 rows: s = s0 + 16 + half*8 + i.
    float v0[8], v1[8];
    float mloc = NEG_INF;
#pragma unroll
    for (int i = 0; i < 8; ++i) {
      int sa = s0 + half * 8 + i;
      int sb = sa + 16;
      v0[i] = mb[sa] ? st0[i] * SCALE : NEG_INF;
      v1[i] = mb[sb] ? st1[i] * SCALE : NEG_INF;
      mloc = fmaxf(mloc, fmaxf(v0[i], v1[i]));
    }
    mloc = fmaxf(mloc, __shfl_xor(mloc, 16, 32));
    float m_new = fmaxf(m_run, mloc);
    float corr = (m_new == NEG_INF) ? 1.f : __expf(m_run - m_new);

    // Build P as an A-fragment: j=i <- st0 values, j=8+i <- st1 values.
    float ssum = 0.f;
    v16h p;
#pragma unroll
    for (int i = 0; i < 8; ++i) {
      float e0 = (v0[i] == NEG_INF) ? 0.f : __expf(v0[i] - m_new);
      float e1 = (v1[i] == NEG_INF) ? 0.f : __expf(v1[i] - m_new);
      p[i]     = (_Float16)e0;
      p[i + 8] = (_Float16)e1;
      ssum += e0 + e1;
    }
    ssum += __shfl_xor(ssum, 16, 32);
    ell = ell * corr + ssum;
    m_run = m_new;

    // Rescale O accumulators: O row t = half*8 + i, factor lives at lane t.
#pragma unroll
    for (int i = 0; i < 8; ++i) {
      float f = __shfl(corr, (half << 3) + i, 32);
      o0[i] *= f; o1[i] *= f; o2[i] *= f; o3[i] *= f;
    }

    // O += P * V  (B-operand from Vt rows d, K = s contiguous).
    o0 = wmma16(p, load_frag(vb, 0,  s0, Tc), o0);
    o1 = wmma16(p, load_frag(vb, 16, s0, Tc), o1);
    o2 = wmma16(p, load_frag(vb, 32, s0, Tc), o2);
    o3 = wmma16(p, load_frag(vb, 48, s0, Tc), o3);
  }

  // Negated softmax: out = -(P @ V) / ell. Write f16 [B*T, C].
  _Float16* ob = Out + (size_t)b * Tc * Cc + h * Dd;
  int col = lane & 15;
#pragma unroll
  for (int i = 0; i < 8; ++i) {
    float li = __shfl(ell, (half << 3) + i, 32);
    float inv = -1.f / li;
    int t = t0 + half * 8 + i;
    ob[(size_t)t * Cc + 0  + col] = (_Float16)(o0[i] * inv);
    ob[(size_t)t * Cc + 16 + col] = (_Float16)(o1[i] * inv);
    ob[(size_t)t * Cc + 32 + col] = (_Float16)(o2[i] * inv);
    ob[(size_t)t * Cc + 48 + col] = (_Float16)(o3[i] * inv);
  }
}

// Mask passthrough (second tuple output) written as 0/1 floats.
__global__ void mask_out_kernel(const unsigned char* __restrict__ mask,
                                float* __restrict__ out, int n) {
  int i = blockIdx.x * blockDim.x + threadIdx.x;
  if (i < n) out[i] = mask[i] ? 1.f : 0.f;
}

// ---------------------------------------------------------------------------
extern "C" void kernel_launch(void* const* d_in, const int* in_sizes, int n_in,
                              void* d_out, int out_size, void* d_ws, size_t ws_size,
                              hipStream_t stream) {
  const float* xkv = (const float*)d_in[0];
  const float* xq  = (const float*)d_in[1];
  const unsigned char* mask = (const unsigned char*)d_in[2];  // jnp bool = 1 byte
  const float* w_qc = (const float*)d_in[3];
  const float* w_kc = (const float*)d_in[4];
  const float* w_vc = (const float*)d_in[5];
  const float* g_q = (const float*)d_in[6];
  const float* b_q = (const float*)d_in[7];
  const float* g_k = (const float*)d_in[8];
  const float* b_k = (const float*)d_in[9];
  const float* g_v = (const float*)d_in[10];
  const float* b_v = (const float*)d_in[11];
  const float* Wq = (const float*)d_in[12];
  const float* bq = (const float*)d_in[13];
  const float* Wk = (const float*)d_in[14];
  const float* bk = (const float*)d_in[15];
  const float* Wv = (const float*)d_in[16];
  const float* bv = (const float*)d_in[17];
  const float* Wp = (const float*)d_in[18];
  const float* bp = (const float*)d_in[19];
  float* out = (float*)d_out;

  const size_t MC = (size_t)Bc * Tc * Cc;  // 4M elements
  const size_t WW = (size_t)Cc * Cc;       // 1M elements
  _Float16* ws   = (_Float16*)d_ws;
  _Float16* qln  = ws;
  _Float16* kln  = qln  + MC;
  _Float16* vln  = kln  + MC;
  _Float16* qpr  = vln  + MC;
  _Float16* kpr  = qpr  + MC;
  _Float16* vprT = kpr  + MC;
  _Float16* att  = vprT + MC;
  _Float16* Wq16 = att  + MC;
  _Float16* Wk16 = Wq16 + WW;
  _Float16* Wv16 = Wk16 + WW;
  _Float16* Wp16 = Wv16 + WW;

  // Stage 0: weights to f16
  {
    int n = (int)WW, blk = 256, grd = (n + blk - 1) / blk;
    f32_to_f16_kernel<<<grd, blk, 0, stream>>>(Wq, Wq16, n);
    f32_to_f16_kernel<<<grd, blk, 0, stream>>>(Wk, Wk16, n);
    f32_to_f16_kernel<<<grd, blk, 0, stream>>>(Wv, Wv16, n);
    f32_to_f16_kernel<<<grd, blk, 0, stream>>>(Wp, Wp16, n);
  }

  // Stage 1: dwconv + mask + LN for q/k/v
  conv_ln_kernel<<<dim3(Bc * Tc, 3), 256, 0, stream>>>(
      xq, xkv, w_qc, w_kc, w_vc, g_q, b_q, g_k, b_k, g_v, b_v, mask,
      qln, kln, vln);

  // Stage 2: projections. M = B*T = 4096, N = K = C = 1024.
  const int M = Bc * Tc;
  dim3 ggrid(Cc / 32, M / (32 * 4));
  gemm_wmma_kernel<<<ggrid, 128, 0, stream>>>(qln, Wq16, bq, M, Cc, Cc, 0,
                                              mask, nullptr, qpr, nullptr);
  gemm_wmma_kernel<<<ggrid, 128, 0, stream>>>(kln, Wk16, bk, M, Cc, Cc, 0,
                                              mask, nullptr, kpr, nullptr);
  gemm_wmma_kernel<<<ggrid, 128, 0, stream>>>(vln, Wv16, bv, M, Cc, Cc, 1,
                                              mask, nullptr, vprT, nullptr);

  // Stage 3: fused attention, one wave per 16 q rows per (b,h)
  attn_wmma_kernel<<<dim3(Tc / 16, Bc * Hh), 32, 0, stream>>>(
      qpr, kpr, vprT, mask, att);

  // Stage 4: output projection + mask + residual -> f32 [B, C, T]
  gemm_wmma_kernel<<<ggrid, 128, 0, stream>>>(att, Wp16, bp, M, Cc, Cc, 2,
                                              mask, xkv, nullptr, out);

  // Tuple output #2: mask passthrough
  mask_out_kernel<<<(Bc * Tc + 255) / 256, 256, 0, stream>>>(
      mask, out + MC, Bc * Tc);
}
